// OptimizedPaCLoss_1090921693583
// MI455X (gfx1250) — matrix operand order repaired and checked by reading, not compiled, not run on hardware
//
#include <hip/hip_runtime.h>
#include <hip/hip_bf16.h>
#include <math.h>

// ---------------------------------------------------------------------------
// PaC triplet loss on MI455X (gfx1250, wave32).
//   1) chol_kernel     : per-batch Cholesky of Sigma + 1e-4 I (f32, in ws)
//   2) prep_a_kernel   : Z1t (128x256) f32 -> f16 (row scaling irrelevant to top-k)
//   3) prep_neg_kernel : negatives row-normalized f32 -> f16 (8192x256)
//   4) sims_wmma_kernel: 128x8192 GEMM (K=256) via v_wmma_f32_16x16x32_f16
//   5) topk_kernel     : exact top-32 per row via LDS argmax extraction
//   6) final_kernel    : per-(b,k): forward substitution L y = (p - x) for the
//                        positive + 32 selected negatives (wave-parallel dots),
//                        semi-hard min, hinge, atomicAdd into scalar out.
// ---------------------------------------------------------------------------

typedef __attribute__((ext_vector_type(16))) _Float16 v16h;
typedef __attribute__((ext_vector_type(8)))  _Float16 v8h;
typedef __attribute__((ext_vector_type(8)))  float    v8f;

#define BDIM 16
#define KDIM 8
#define DDIM 256
#define NDIM 8192
#define BK   128           // B*K rows
#define TOPM 32
#define ALPHA_C   0.2f
#define BASE_REG_C 1e-4f

__global__ void zero_kernel(float* out) {
    if (threadIdx.x == 0) out[0] = 0.0f;
}

// ---- 1) Cholesky (right-looking, one block of 256 threads per batch) -------
__global__ void chol_kernel(const float* __restrict__ Sigma,
                            float* __restrict__ Lmat) {
    const int b = blockIdx.x;
    const float* S = Sigma + (size_t)b * DDIM * DDIM;
    float* A = Lmat + (size_t)b * DDIM * DDIM;

    for (int idx = threadIdx.x; idx < DDIM * DDIM; idx += blockDim.x) {
        const int i = idx >> 8, j = idx & (DDIM - 1);
        float v = S[idx];
        if (i == j) v += BASE_REG_C;
        A[idx] = (j <= i) ? v : 0.0f;   // keep lower triangle, zero upper
    }
    __syncthreads();

    for (int kk = 0; kk < DDIM; ++kk) {
        if (threadIdx.x == 0) A[kk * DDIM + kk] = sqrtf(A[kk * DDIM + kk]);
        __syncthreads();
        const float dk = A[kk * DDIM + kk];
        for (int i = kk + 1 + threadIdx.x; i < DDIM; i += blockDim.x)
            A[i * DDIM + kk] /= dk;
        __syncthreads();
        // trailing rank-1 update: writes only columns > kk, reads only col kk
        for (int i = kk + 1; i < DDIM; ++i) {
            const float lik = A[i * DDIM + kk];
            for (int j = kk + 1 + threadIdx.x; j <= i; j += blockDim.x)
                A[i * DDIM + j] -= lik * A[j * DDIM + kk];
        }
        __syncthreads();
    }
}

// ---- 2) Z1 -> f16 ----------------------------------------------------------
__global__ void prep_a_kernel(const float* __restrict__ Z1,
                              _Float16* __restrict__ A16) {
    const int idx = blockIdx.x * blockDim.x + threadIdx.x;
    if (idx < BK * DDIM) A16[idx] = (_Float16)Z1[idx];
}

// ---- 3) normalize negatives rows -> f16 ------------------------------------
__global__ void prep_neg_kernel(const float* __restrict__ negs,
                                _Float16* __restrict__ N16) {
    __shared__ float red[256];
    const int n = blockIdx.x;
    const float v = negs[(size_t)n * DDIM + threadIdx.x];
    red[threadIdx.x] = v * v;
    __syncthreads();
    for (int s = 128; s > 0; s >>= 1) {
        if (threadIdx.x < s) red[threadIdx.x] += red[threadIdx.x + s];
        __syncthreads();
    }
    const float inv = rsqrtf(red[0]);
    N16[(size_t)n * DDIM + threadIdx.x] = (_Float16)(v * inv);
}

// ---- 4) sims = A16 (128x256) * N16^T (256x8192) via WMMA -------------------
// One wave per 16x16 output tile. ISA 7.12.2 layouts:
//   A 16x32 f16 : lane<16 holds M=lane, K-blocks {k0+0..7, k0+16..23};
//                 lane>=16 holds M=lane-16, K-blocks {k0+8..15, k0+24..31}
//   B 32x16 f16 : lane l holds N=l&15, K-blocks same as A (k0+8*(l>>4)+0..7, k0+16+8*(l>>4)+0..7)
//   C/D f32     : VGPR r -> M = r + 8*(lane>>4), N = lane&15
__global__ void sims_wmma_kernel(const _Float16* __restrict__ A16,
                                 const _Float16* __restrict__ N16,
                                 float* __restrict__ sims) {
    const int lane = threadIdx.x;        // 0..31, whole wave active (EXEC all 1s)
    const int hb   = lane >> 4;          // half of wave
    const int l16  = lane & 15;
    const int n0 = blockIdx.x * 16;
    const int m0 = blockIdx.y * 16;
    const int m  = m0 + l16;
    const int n  = n0 + l16;

    v8f acc = {};
    for (int k0 = 0; k0 < DDIM; k0 += 32) {
        const v8h* arow = (const v8h*)(A16 + (size_t)m * DDIM + k0);
        const v8h alo = arow[hb];        // K offset 8*hb
        const v8h ahi = arow[2 + hb];    // K offset 16 + 8*hb
        v16h a;
        #pragma unroll
        for (int e = 0; e < 8; ++e) { a[e] = alo[e]; a[8 + e] = ahi[e]; }

        const v8h* brow = (const v8h*)(N16 + (size_t)n * DDIM + k0);
        const v8h blo = brow[hb];
        const v8h bhi = brow[2 + hb];
        v16h bv;
        #pragma unroll
        for (int e = 0; e < 8; ++e) { bv[e] = blo[e]; bv[8 + e] = bhi[e]; }

        acc = __builtin_amdgcn_wmma_f32_16x16x32_f16(
            /*neg_a=*/false, a, /*neg_b=*/false, bv,
            /*c_mod=*/(short)0, acc, /*reuse_a=*/false, /*reuse_b=*/false);
    }
    #pragma unroll
    for (int r = 0; r < 8; ++r)
        sims[(size_t)(m0 + r + 8 * hb) * NDIM + n0 + l16] = acc[r];
}

// ---- 5) exact top-32 per row (row cached in 32KB LDS) ----------------------
__global__ void topk_kernel(const float* __restrict__ sims,
                            int* __restrict__ sel) {
    __shared__ float row[NDIM];
    __shared__ float rv[256];
    __shared__ int   ri[256];
    const int r = blockIdx.x;
    for (int j = threadIdx.x; j < NDIM; j += 256)
        row[j] = sims[(size_t)r * NDIM + j];
    __syncthreads();

    for (int it = 0; it < TOPM; ++it) {
        float best = -INFINITY; int bi = 0;
        for (int j = threadIdx.x; j < NDIM; j += 256) {
            const float v = row[j];
            if (v > best) { best = v; bi = j; }
        }
        rv[threadIdx.x] = best; ri[threadIdx.x] = bi;
        __syncthreads();
        for (int s = 128; s > 0; s >>= 1) {
            if (threadIdx.x < s && rv[threadIdx.x + s] > rv[threadIdx.x]) {
                rv[threadIdx.x] = rv[threadIdx.x + s];
                ri[threadIdx.x] = ri[threadIdx.x + s];
            }
            __syncthreads();
        }
        if (threadIdx.x == 0) { sel[r * TOPM + it] = ri[0]; row[ri[0]] = -INFINITY; }
        __syncthreads();
    }
}

// ---- 6) whiten-by-solve + semi-hard hinge ----------------------------------
// One block per (b,k); 8 waves each own 4-5 of the 33 solves (32 negs + 1 pos).
// Forward substitution: y distributed across lanes (y[j] on lane j&31, slot j>>5),
// per-row dot reduced with a 5-step shfl_xor butterfly.
__global__ void final_kernel(const float* __restrict__ Z1,
                             const float* __restrict__ Z2,
                             const int* __restrict__ matching,
                             const float* __restrict__ negs,
                             const float* __restrict__ Lmat,
                             const int* __restrict__ sel,
                             float* __restrict__ out) {
    __shared__ float sdist[33];
    const int bk   = blockIdx.x;        // 0..127
    const int b    = bk >> 3;
    const int wave = threadIdx.x >> 5;
    const int lane = threadIdx.x & 31;
    const float* Lb   = Lmat + (size_t)b * DDIM * DDIM;
    const float* arow = Z1 + (size_t)bk * DDIM;

    for (int m = wave; m < 33; m += 8) {
        const float* crow;
        if (m == 32) {
            const int mk = matching[bk];
            crow = Z2 + ((size_t)b * KDIM + mk) * DDIM;
        } else {
            const int ni = sel[bk * TOPM + m];
            crow = negs + (size_t)ni * DDIM;
        }
        float y[8];
        #pragma unroll
        for (int r = 0; r < 8; ++r) y[r] = 0.0f;
        float norm2 = 0.0f;

        for (int i = 0; i < DDIM; ++i) {
            const float* Lrow = Lb + (size_t)i * DDIM;
            float partial = 0.0f;
            #pragma unroll
            for (int r = 0; r < 8; ++r) {
                const int j = r * 32 + lane;
                if (j < i) partial += Lrow[j] * y[r];
            }
            #pragma unroll
            for (int off = 16; off > 0; off >>= 1)
                partial += __shfl_xor(partial, off, 32);
            const float vi = arow[i] - crow[i];
            const float yi = (vi - partial) / Lrow[i];   // Lrow[i] == L[i][i]
            if ((i & 31) == lane) y[i >> 5] = yi;
            norm2 += yi * yi;                            // identical on all lanes
        }
        if (lane == 0) sdist[m] = sqrtf(norm2);
    }
    __syncthreads();

    if (threadIdx.x == 0) {
        const float dp = sdist[32];
        float amin = INFINITY, vmin = INFINITY;
        int any = 0;
        for (int m = 0; m < TOPM; ++m) {
            const float dn = sdist[m];
            amin = fminf(amin, dn);
            if (dn > dp) { any = 1; vmin = fminf(vmin, dn); }
        }
        const float dneg = any ? vmin : amin;
        const float loss = fmaxf(dp - dneg + ALPHA_C, 0.0f);
        atomicAdd(out, loss * (1.0f / (float)BK));
    }
}

// ---------------------------------------------------------------------------
extern "C" void kernel_launch(void* const* d_in, const int* in_sizes, int n_in,
                              void* d_out, int out_size, void* d_ws, size_t ws_size,
                              hipStream_t stream) {
    (void)in_sizes; (void)n_in; (void)out_size; (void)ws_size;

    const float* Z1       = (const float*)d_in[0];   // (16,8,256)
    const float* Z2       = (const float*)d_in[1];   // (16,8,256)
    const int*   matching = (const int*)  d_in[2];   // (16,8)
    const float* negs     = (const float*)d_in[3];   // (8192,256)
    const float* Sigma    = (const float*)d_in[4];   // (16,256,256)
    float* out = (float*)d_out;

    // workspace carve-out (all offsets 64KB-multiples -> vector-load aligned)
    char* ws = (char*)d_ws;
    float*    Lmat = (float*)ws;    ws += (size_t)BDIM * DDIM * DDIM * sizeof(float);    // 4 MB
    _Float16* A16  = (_Float16*)ws; ws += (size_t)BK * DDIM * sizeof(_Float16);          // 64 KB
    _Float16* N16  = (_Float16*)ws; ws += (size_t)NDIM * DDIM * sizeof(_Float16);        // 4 MB
    float*    sims = (float*)ws;    ws += (size_t)BK * NDIM * sizeof(float);             // 4 MB
    int*      sel  = (int*)ws;      ws += (size_t)BK * TOPM * sizeof(int);               // 16 KB

    zero_kernel<<<1, 32, 0, stream>>>(out);
    chol_kernel<<<BDIM, 256, 0, stream>>>(Sigma, Lmat);
    prep_a_kernel<<<(BK * DDIM + 255) / 256, 256, 0, stream>>>(Z1, A16);
    prep_neg_kernel<<<NDIM, 256, 0, stream>>>(negs, N16);
    sims_wmma_kernel<<<dim3(NDIM / 16, BK / 16), 32, 0, stream>>>(A16, N16, sims);
    topk_kernel<<<BK, 256, 0, stream>>>(sims, sel);
    final_kernel<<<BK, 256, 0, stream>>>(Z1, Z2, matching, negs, Lmat, sel, out);
}
